// WindowAttention_4715874091414
// MI455X (gfx1250) — matrix operand order repaired and checked by reading, not compiled, not run on hardware
//
#include <hip/hip_runtime.h>
#include <hip/hip_fp16.h>
#include <math.h>

typedef __attribute__((ext_vector_type(16))) _Float16 v16h;
typedef __attribute__((ext_vector_type(8)))  _Float16 v8h;
typedef __attribute__((ext_vector_type(8)))  float    v8f;

// Problem constants
#define B_   4
#define P_   4096
#define F_   256
#define H_   8
#define DH_  32
#define WSZ_ 64
#define NEWP_ 64
#define U_   25
#define INNER_ 256   // H*Dh
#define NQKV_ 768    // 3*inner

// ---------------------------------------------------------------------------
// CDNA5 async global->LDS copy helpers (ASYNCcnt-tracked DMA path)
// ---------------------------------------------------------------------------
__device__ inline uint32_t lds_off32(const void* p) {
  // gfx1250 flat LDS aperture keeps the LDS byte offset in addr[31:0].
  return (uint32_t)(uintptr_t)p;
}

__device__ inline void async_cp16(const void* g, const void* l) {
  asm volatile("global_load_async_to_lds_b128 %0, %1, off"
               :: "v"(lds_off32(l)), "v"(g)
               : "memory");
}

__device__ inline void wait_async0() {
  asm volatile("s_wait_asynccnt 0" ::: "memory");
}

// ---------------------------------------------------------------------------
// WMMA GEMM core: C(128x64) += A(128xK) * B(Kx64).
// Staging converts f32->f16 ONCE into fragment-native LDS layouts:
//   Ash[r][k]  row-major f16  -> lane A-frag = two contiguous v8h loads
//   Bsh[c][k]  K-major   f16  -> lane B-frag = one contiguous v16h load
// Per-row gate (x*y fusion) applied during the staging convert via ysh.
// 8 waves/block; wave w owns rows [w*16, w*16+16), all 64 cols (4 accums).
// ---------------------------------------------------------------------------
__device__ inline void gemm_tile(const float* __restrict__ A, int lda,
                                 const float* __restrict__ Bm, int ldb,
                                 int m0, int n0, int K,
                                 const float* __restrict__ gatev,  // y+m0 or null
                                 _Float16* Ash /*[128*32]*/,
                                 _Float16* Bsh /*[64*32]*/,
                                 float* ysh /*[128]*/, v8f acc[4]) {
  const int tid  = threadIdx.x;
  const int lane = tid & 31;
  const int wave = tid >> 5;

  const int ar  = lane & 15;          // A row within wave strip
  const int akb = (lane >> 4) << 3;   // A K-half base (0 or 8)
  const int bn  = lane & 15;          // B column within 16-tile
  const int bkb = (lane >> 4) << 4;   // B K base (0 or 16)

  if (tid < 128) ysh[tid] = gatev ? gatev[tid] : 1.0f;
  __syncthreads();

  for (int k0 = 0; k0 < K; k0 += 32) {
    // --- stage A tile: 128 rows x 32 f32, gate + cvt, 1024 float4 segs ---
    for (int s = tid; s < 1024; s += 256) {
      int r = s >> 3, c = (s & 7) << 2;
      const float4 w = *(const float4*)(A + (size_t)(m0 + r) * lda + k0 + c);
      const float g = ysh[r];
      _Float16* dst = Ash + r * 32 + c;
      dst[0] = (_Float16)(w.x * g);
      dst[1] = (_Float16)(w.y * g);
      dst[2] = (_Float16)(w.z * g);
      dst[3] = (_Float16)(w.w * g);
    }
    // --- stage B tile: 32 rows x 64 f32 -> transposed f16 [c][k] ---
    for (int s = tid; s < 512; s += 256) {
      int k = s >> 4, c0 = (s & 15) << 2;
      const float4 w = *(const float4*)(Bm + (size_t)(k0 + k) * ldb + n0 + c0);
      Bsh[(c0 + 0) * 32 + k] = (_Float16)w.x;
      Bsh[(c0 + 1) * 32 + k] = (_Float16)w.y;
      Bsh[(c0 + 2) * 32 + k] = (_Float16)w.z;
      Bsh[(c0 + 3) * 32 + k] = (_Float16)w.w;
    }
    __syncthreads();

    // --- A fragment: two contiguous 16B LDS loads, no conversion ---
    const _Float16* Aw = Ash + (wave * 16 + ar) * 32;
    v8h alo = *(const v8h*)(Aw + bkb);
    v8h ahi = *(const v8h*)(Aw + bkb + 8);
    v16h af = __builtin_shufflevector(alo, ahi, 0, 1, 2, 3, 4, 5, 6, 7, 8, 9,
                                      10, 11, 12, 13, 14, 15);

    // --- B fragments: one contiguous 32B LDS load each; preload all 4 ---
    v16h bf0 = *(const v16h*)(Bsh + (0 * 16 + bn) * 32 + bkb);
    v16h bf1 = *(const v16h*)(Bsh + (1 * 16 + bn) * 32 + bkb);
    v16h bf2 = *(const v16h*)(Bsh + (2 * 16 + bn) * 32 + bkb);
    v16h bf3 = *(const v16h*)(Bsh + (3 * 16 + bn) * 32 + bkb);

    acc[0] = __builtin_amdgcn_wmma_f32_16x16x32_f16(false, af, false, bf0,
                                                    (short)0, acc[0], false, false);
    acc[1] = __builtin_amdgcn_wmma_f32_16x16x32_f16(false, af, false, bf1,
                                                    (short)0, acc[1], false, false);
    acc[2] = __builtin_amdgcn_wmma_f32_16x16x32_f16(false, af, false, bf2,
                                                    (short)0, acc[2], false, false);
    acc[3] = __builtin_amdgcn_wmma_f32_16x16x32_f16(false, af, false, bf3,
                                                    (short)0, acc[3], false, false);
    __syncthreads();
  }
}

// ---------------------------------------------------------------------------
// Kernel 1: row means of x -> xm (B*P)
// ---------------------------------------------------------------------------
__global__ __launch_bounds__(256) void mean_kernel(const float* __restrict__ x,
                                                   float* __restrict__ xm) {
  int row  = blockIdx.x * 8 + (threadIdx.x >> 5);
  int lane = threadIdx.x & 31;
  const float* xr = x + (size_t)row * F_;
  float s = 0.f;
#pragma unroll
  for (int i = 0; i < F_ / 32; ++i) s += xr[lane + i * 32];
#pragma unroll
  for (int off = 16; off > 0; off >>= 1) s += __shfl_xor(s, off, 32);
  if (lane == 0) xm[row] = s * (1.0f / F_);
}

// ---------------------------------------------------------------------------
// Kernel 2: y[b,j] = relu(sum_p xm[b,p]*Wwa[p,j] + bwa[j])  (streams 64MB Wwa)
// ---------------------------------------------------------------------------
__global__ __launch_bounds__(256) void wa_gemv_kernel(const float* __restrict__ xm,
                                                      const float* __restrict__ Wwa,
                                                      const float* __restrict__ bwa,
                                                      float* __restrict__ y) {
  int j = blockIdx.x * 256 + threadIdx.x;
  float a0 = 0.f, a1 = 0.f, a2 = 0.f, a3 = 0.f;
#pragma unroll 4
  for (int p = 0; p < P_; ++p) {
    if ((p & 63) == 0 && p + 64 < P_)
      __builtin_prefetch(&Wwa[(size_t)(p + 64) * P_ + j], 0, 0);
    float w = Wwa[(size_t)p * P_ + j];
    a0 = fmaf(xm[p], w, a0);
    a1 = fmaf(xm[P_ + p], w, a1);
    a2 = fmaf(xm[2 * P_ + p], w, a2);
    a3 = fmaf(xm[3 * P_ + p], w, a3);
  }
  float bb = bwa[j];
  y[j]          = fmaxf(a0 + bb, 0.f);
  y[P_ + j]     = fmaxf(a1 + bb, 0.f);
  y[2 * P_ + j] = fmaxf(a2 + bb, 0.f);
  y[3 * P_ + j] = fmaxf(a3 + bb, 0.f);
}

// ---------------------------------------------------------------------------
// Kernel 3: qkv = (x * y) @ Wqkv, scattered to windowed layout [b][h][n][w][d]
// (p = w*NEWP + n per the reference reshape)
// ---------------------------------------------------------------------------
__global__ __launch_bounds__(256) void qkv_gemm_kernel(
    const float* __restrict__ x, const float* __restrict__ y,
    const float* __restrict__ Wqkv, float* __restrict__ qw,
    float* __restrict__ kw, float* __restrict__ vw) {
  __shared__ __align__(32) _Float16 Ash[128 * 32];
  __shared__ __align__(32) _Float16 Bsh[64 * 32];
  __shared__ float ysh[128];
  int m0 = blockIdx.y * 128, n0 = blockIdx.x * 64;
  int lane = threadIdx.x & 31, wave = threadIdx.x >> 5;

  v8f z = {0.f, 0.f, 0.f, 0.f, 0.f, 0.f, 0.f, 0.f};
  v8f acc[4] = {z, z, z, z};

  gemm_tile(x, F_, Wqkv, NQKV_, m0, n0, F_, y + m0, Ash, Bsh, ysh, acc);

#pragma unroll
  for (int nt = 0; nt < 4; ++nt) {
#pragma unroll
    for (int v = 0; v < 8; ++v) {
      int row = m0 + wave * 16 + ((lane >> 4) << 3) + v;  // b*P + p
      int col = n0 + nt * 16 + (lane & 15);               // 0..767
      int b = row >> 12, p = row & (P_ - 1);
      int w = p >> 6, n = p & 63;                         // p = w*64 + n
      int mat = col >> 8, cc = col & 255, h = cc >> 5, d = cc & 31;
      float* dst = (mat == 0) ? qw : (mat == 1) ? kw : vw;
      dst[((((size_t)b * H_ + h) * NEWP_ + n) * WSZ_ + w) * DH_ + d] =
          acc[nt][v];
    }
  }
}

// ---------------------------------------------------------------------------
// Kernel 4: sampled-top-k window attention, one block per (b,h,window n).
// q/k/v/index tiles staged with CDNA5 async global->LDS DMA (ASYNCcnt).
// ---------------------------------------------------------------------------
__global__ __launch_bounds__(128) void attn_kernel(
    const float* __restrict__ qw, const float* __restrict__ kw,
    const float* __restrict__ vw, const float* __restrict__ pos_emb,
    const int* __restrict__ index_sample, float* __restrict__ ctx) {
  __shared__ __align__(16) float qs[WSZ_ * DH_];
  __shared__ __align__(16) float ks_[WSZ_ * DH_];
  __shared__ __align__(16) float vs[WSZ_ * DH_];
  __shared__ __align__(16) int   idxs[WSZ_ * U_];
  __shared__ float Msh[WSZ_];
  __shared__ int   topIdx[U_];
  __shared__ int   wslot[WSZ_];
  __shared__ float att[U_ * WSZ_];
  __shared__ float upd[U_ * DH_];
  __shared__ float vmean[DH_];

  int unit = blockIdx.x;        // ((b*H + h)*NEWP + n)
  int n = unit & 63;
  int bh = unit >> 6;
  int h = bh & 7, b = bh >> 3;
  int tid = threadIdx.x;

  const size_t base = (size_t)unit * (WSZ_ * DH_);
  // Async DMA: 16B per lane-op, 2KB (512-float) stride across the 128-thread block.
  for (int i = (tid << 2); i < WSZ_ * DH_; i += 512) {
    async_cp16(qw + base + i, &qs[i]);
    async_cp16(kw + base + i, &ks_[i]);
    async_cp16(vw + base + i, &vs[i]);
  }
  for (int i = (tid << 2); i < WSZ_ * U_; i += 512)
    async_cp16(index_sample + i, &idxs[i]);
  wait_async0();
  __syncthreads();

  // Sampled QK -> sparsity measure M[i] = max_s - sum_s / Wsz
  if (tid < WSZ_) {
    int i = tid;
    float qreg[DH_];
#pragma unroll
    for (int d = 0; d < DH_; ++d) qreg[d] = qs[i * DH_ + d];
    float mx = -INFINITY, sm = 0.f;
    for (int s = 0; s < U_; ++s) {
      int ki = idxs[i * U_ + s];
      float dot = 0.f;
#pragma unroll
      for (int d = 0; d < DH_; ++d) dot = fmaf(qreg[d], ks_[ki * DH_ + d], dot);
      mx = fmaxf(mx, dot);
      sm += dot;
    }
    Msh[i] = mx - sm * (1.0f / WSZ_);
  }
  __syncthreads();

  // Stable top-U via rank (matches jax.lax.top_k descending, ties by index)
  if (tid < WSZ_) {
    int i = tid;
    float mi = Msh[i];
    int rank = 0;
    for (int j = 0; j < WSZ_; ++j) {
      float mj = Msh[j];
      rank += (mj > mi) || (mj == mi && j < i);
    }
    if (rank < U_) topIdx[rank] = i;
    wslot[i] = -1;
  }
  __syncthreads();
  if (tid < U_) wslot[topIdx[tid]] = tid;
  __syncthreads();

  // scores[u][j] = q[topIdx[u]] . k[j] * scale + pos_emb[j - i + Wsz - 1]
  const float scale = 0.1767766952966369f;  // 1/sqrt(32)
  for (int t = tid; t < U_ * WSZ_; t += 128) {
    int u = t >> 6, j = t & 63;
    int i = topIdx[u];
    float dot = 0.f;
#pragma unroll
    for (int d = 0; d < DH_; ++d) dot = fmaf(qs[i * DH_ + d], ks_[j * DH_ + d], dot);
    att[u * WSZ_ + j] = dot * scale + pos_emb[j - i + WSZ_ - 1];
  }
  __syncthreads();

  // softmax over j (threads 0..24) and v-mean (threads 32..63), disjoint sets
  if (tid < U_) {
    float mx = -INFINITY;
    for (int j = 0; j < WSZ_; ++j) mx = fmaxf(mx, att[tid * WSZ_ + j]);
    float s = 0.f;
    for (int j = 0; j < WSZ_; ++j) {
      float e = expf(att[tid * WSZ_ + j] - mx);
      att[tid * WSZ_ + j] = e;
      s += e;
    }
    float inv = 1.0f / s;
    for (int j = 0; j < WSZ_; ++j) att[tid * WSZ_ + j] *= inv;
  }
  if (tid >= 32 && tid < 64) {
    int d = tid - 32;
    float s = 0.f;
    for (int w = 0; w < WSZ_; ++w) s += vs[w * DH_ + d];
    vmean[d] = s * (1.0f / WSZ_);
  }
  __syncthreads();

  // upd[u][d] = sum_j attn[u][j] * v[j][d]
  for (int t = tid; t < U_ * DH_; t += 128) {
    int u = t >> 5, d = t & 31;
    float s = 0.f;
    for (int j = 0; j < WSZ_; ++j) s = fmaf(att[u * WSZ_ + j], vs[j * DH_ + d], s);
    upd[u * DH_ + d] = s;
  }
  __syncthreads();

  // scatter into ctx at position p' = n*Wsz + w, feature h*Dh + d
  float* cbase = ctx + ((size_t)b * P_ + (size_t)n * WSZ_) * INNER_ + h * DH_;
  for (int t = tid; t < WSZ_ * DH_; t += 128) {
    int w = t >> 5, d = t & 31;
    int sl = wslot[w];
    float val = (sl >= 0) ? upd[sl * DH_ + d] : vmean[d];
    cbase[(size_t)w * INNER_ + d] = val;
  }
}

// ---------------------------------------------------------------------------
// Kernel 5: out = ctx @ Wout + bout  (WMMA, f32 accumulate)
// ---------------------------------------------------------------------------
__global__ __launch_bounds__(256) void out_gemm_kernel(
    const float* __restrict__ ctx, const float* __restrict__ Wout,
    const float* __restrict__ bout, float* __restrict__ out) {
  __shared__ __align__(32) _Float16 Ash[128 * 32];
  __shared__ __align__(32) _Float16 Bsh[64 * 32];
  __shared__ float ysh[128];
  int m0 = blockIdx.y * 128, n0 = blockIdx.x * 64;
  int lane = threadIdx.x & 31, wave = threadIdx.x >> 5;

  v8f z = {0.f, 0.f, 0.f, 0.f, 0.f, 0.f, 0.f, 0.f};
  v8f acc[4] = {z, z, z, z};

  gemm_tile(ctx, INNER_, Wout, F_, m0, n0, INNER_, nullptr, Ash, Bsh, ysh, acc);

#pragma unroll
  for (int nt = 0; nt < 4; ++nt) {
    int col = n0 + nt * 16 + (lane & 15);
    float bb = bout[col];
#pragma unroll
    for (int v = 0; v < 8; ++v) {
      int row = m0 + wave * 16 + ((lane >> 4) << 3) + v;
      out[(size_t)row * F_ + col] = acc[nt][v] + bb;
    }
  }
}

// ---------------------------------------------------------------------------
extern "C" void kernel_launch(void* const* d_in, const int* in_sizes, int n_in,
                              void* d_out, int out_size, void* d_ws,
                              size_t ws_size, hipStream_t stream) {
  const float* x        = (const float*)d_in[0];
  const float* Wqkv     = (const float*)d_in[1];
  const float* pos_emb  = (const float*)d_in[2];
  const float* Wwa      = (const float*)d_in[3];
  const float* bwa      = (const float*)d_in[4];
  const float* Wout     = (const float*)d_in[5];
  const float* bout     = (const float*)d_in[6];
  const int*   idx      = (const int*)d_in[7];
  float* out = (float*)d_out;

  // workspace layout (floats): xm[16384] | y[16384] | qw | kw | vw | ctx
  float* ws  = (float*)d_ws;
  const size_t NWIN = (size_t)B_ * H_ * NEWP_ * WSZ_ * DH_;  // 4,194,304
  float* xm  = ws;
  float* yv  = xm + (size_t)B_ * P_;
  float* qw  = yv + (size_t)B_ * P_;
  float* kw  = qw + NWIN;
  float* vw  = kw + NWIN;
  float* ctx = vw + NWIN;

  mean_kernel<<<(B_ * P_) / 8, 256, 0, stream>>>(x, xm);
  wa_gemv_kernel<<<P_ / 256, 256, 0, stream>>>(xm, Wwa, bwa, yv);
  qkv_gemm_kernel<<<dim3(NQKV_ / 64, (B_ * P_) / 128), 256, 0, stream>>>(
      x, yv, Wqkv, qw, kw, vw);
  attn_kernel<<<B_ * H_ * NEWP_, 128, 0, stream>>>(qw, kw, vw, pos_emb, idx, ctx);
  out_gemm_kernel<<<dim3(F_ / 64, (B_ * P_) / 128), 256, 0, stream>>>(
      ctx, Wout, bout, out);
}
